// SelfAttention_16398185136509
// MI455X (gfx1250) — matrix operand order repaired and checked by reading, not compiled, not run on hardware
//
#include <hip/hip_runtime.h>

// ---------------------------------------------------------------------------
// Self-attention + GLU + LayerNorm for MI455X (gfx1250), wave32 + WMMA f16.
// GEMMs: 128x128 block tile, 32M x 64N per wave, double-buffered async
// global->LDS staging (ASYNCcnt partial waits for overlap).
// ---------------------------------------------------------------------------

typedef _Float16 h8  __attribute__((ext_vector_type(8)));
typedef _Float16 h16 __attribute__((ext_vector_type(16)));
typedef float    f8  __attribute__((ext_vector_type(8)));
typedef float    f4  __attribute__((ext_vector_type(4)));

#define GTM 128         // gemm block tile M
#define GTN 128         // gemm block tile N
#define TK  32          // K step (matches 16x16x32 WMMA)
#define AST 40          // LDS row stride (halves), keeps 16B alignment (80B rows)

// fused kernel tile
#define FTM 128
#define FTN 64

// ---------------- CDNA5 async global->LDS copy (ASYNCcnt tracked) -----------
__device__ inline void async_copy_b128(const _Float16* gptr, _Float16* lptr) {
  unsigned lds_off = (unsigned)(unsigned long long)(const void*)lptr;
  unsigned long long ga = (unsigned long long)gptr;
  asm volatile("global_load_async_to_lds_b128 %0, %1, off"
               :: "v"(lds_off), "v"(ga) : "memory");
}
__device__ inline void async_wait0() {
  asm volatile("s_wait_asynccnt 0x0" ::: "memory");
}
__device__ inline void async_wait4() {
  asm volatile("s_wait_asynccnt 0x4" ::: "memory");
}

// ---------------- fragment builders (CDNA5 16-bit WMMA lane layouts) --------

// A (16x32, MxK): lanes 0-15 row M=lane, K={0..7,16..23}; lanes 16-31 K={8..15,24..31}
__device__ inline h16 make_a_frag(const _Float16* __restrict__ As, int rowBase, int lane) {
  int r   = rowBase + (lane & 15);
  int grp = lane >> 4;
  const _Float16* p = As + r * AST;
  h8 lo = *(const h8*)(p + grp * 8);
  h8 hi = *(const h8*)(p + 16 + grp * 8);
  h16 a;
#pragma unroll
  for (int i = 0; i < 8; ++i) { a[i] = lo[i]; a[i + 8] = hi[i]; }
  return a;
}

// B (32x16, KxN) from LDS tile stored transposed Bs[n][k]:
// lanes 0-15 col N=lane, K={0..7,16..23}; lanes 16-31 K={8..15,24..31}
__device__ inline h16 make_b_frag(const _Float16* __restrict__ Bs, int nBase, int lane) {
  int n  = nBase + (lane & 15);
  int kc = (lane >> 4) * 8;
  const _Float16* p = Bs + n * AST + kc;
  h8 b0 = *(const h8*)(p);
  h8 b1 = *(const h8*)(p + 16);
  h16 b;
#pragma unroll
  for (int i = 0; i < 8; ++i) { b[i] = b0[i]; b[i + 8] = b1[i]; }
  return b;
}

// ---------------- generic WMMA GEMM: C = scale*(A @ Bt^T) + bias ------------
// A: [M x K] row-major (f16), Bt: [N x K] row-major (f16).
// OUTMODE: 0 = f16 C[m*ldc+n], 1 = f16 C[n*ldc+m] (transposed), 2 = f32.
// 8 waves: 4 along M (32 rows each) x 2 along N (64 cols each).
template <int OUTMODE, bool HAS_BIAS>
__global__ __launch_bounds__(256)
void gemm_wmma_kernel(const _Float16* __restrict__ A, int lda, long long sA,
                      const _Float16* __restrict__ Bt, int ldb, long long sB,
                      void* __restrict__ C, int ldc, long long sC,
                      const float* __restrict__ bias, float scale, int K) {
  __shared__ __align__(16) _Float16 As[2][GTM * AST];
  __shared__ __align__(16) _Float16 Bs[2][GTN * AST];

  const int tid  = threadIdx.x;
  const int wave = tid >> 5;
  const int lane = tid & 31;
  const int wm   = (wave & 3) * 32;   // wave M offset in tile
  const int wn   = (wave >> 2) * 64;  // wave N offset in tile
  const int bm   = blockIdx.x * GTM;
  const int bn   = blockIdx.y * GTN;
  const int bz   = blockIdx.z;

  A  += (long long)bz * sA;
  Bt += (long long)bz * sB;

  f8 acc[2][4];
#pragma unroll
  for (int mi = 0; mi < 2; ++mi)
#pragma unroll
    for (int j = 0; j < 4; ++j)
#pragma unroll
      for (int i = 0; i < 8; ++i) acc[mi][j][i] = 0.0f;

  // staging map: 2 threads per 32-wide row, 16 halves (2 x b128) each
  const int srow = tid >> 1;          // 0..127
  const int scol = (tid & 1) * 16;    // 0 / 16
  const _Float16* agBase = A  + (long long)(bm + srow) * lda + scol;
  const _Float16* bgBase = Bt + (long long)(bn + srow) * ldb + scol;
  const int sOff = srow * AST + scol;

  // prologue: stage tile 0 into buffer 0
  async_copy_b128(agBase,     As[0] + sOff);
  async_copy_b128(agBase + 8, As[0] + sOff + 8);
  async_copy_b128(bgBase,     Bs[0] + sOff);
  async_copy_b128(bgBase + 8, Bs[0] + sOff + 8);

  const int nk = K / TK;
  for (int t = 0; t < nk; ++t) {
    const int buf = t & 1;
    if (t + 1 < nk) {
      const int k1 = (t + 1) * TK;
      async_copy_b128(agBase + k1,     As[buf ^ 1] + sOff);
      async_copy_b128(agBase + k1 + 8, As[buf ^ 1] + sOff + 8);
      async_copy_b128(bgBase + k1,     Bs[buf ^ 1] + sOff);
      async_copy_b128(bgBase + k1 + 8, Bs[buf ^ 1] + sOff + 8);
      async_wait4();                    // tile t landed; t+1 still in flight
    } else {
      async_wait0();
    }
    __syncthreads();

    h16 a0 = make_a_frag(As[buf], wm,      lane);
    h16 a1 = make_a_frag(As[buf], wm + 16, lane);
#pragma unroll
    for (int j = 0; j < 4; ++j) {
      h16 bfrag = make_b_frag(Bs[buf], wn + j * 16, lane);
      acc[0][j] = __builtin_amdgcn_wmma_f32_16x16x32_f16(
          false, a0, false, bfrag, (short)0, acc[0][j], false, false);
      acc[1][j] = __builtin_amdgcn_wmma_f32_16x16x32_f16(
          false, a1, false, bfrag, (short)0, acc[1][j], false, false);
    }
    __syncthreads();
  }

  // epilogue: C/D layout -> lane L: n = L&15, m = (L>>4)*8 + vgpr
  const int colLane = lane & 15;
  const int mbase   = (lane >> 4) * 8;
#pragma unroll
  for (int mi = 0; mi < 2; ++mi) {
#pragma unroll
    for (int j = 0; j < 4; ++j) {
      const int n = bn + wn + j * 16 + colLane;
      float bvv = 0.0f;
      if (HAS_BIAS) bvv = bias[n];
#pragma unroll
      for (int g = 0; g < 8; ++g) {
        const int m = bm + wm + mi * 16 + mbase + g;
        const float v = acc[mi][j][g] * scale + bvv;
        if (OUTMODE == 0) {
          _Float16* Cp = (_Float16*)C + (long long)bz * sC;
          Cp[(long long)m * ldc + n] = (_Float16)v;
        } else if (OUTMODE == 1) {
          _Float16* Cp = (_Float16*)C + (long long)bz * sC;
          Cp[(long long)n * ldc + m] = (_Float16)v;
        } else {
          float* Cp = (float*)C + (long long)bz * sC;
          Cp[(long long)m * ldc + n] = v;
        }
      }
    }
  }
}

// ---------------- fused final stage: out@Wo + GLU(q) -> fp32 residual -------
// res = (Ao@Wo + bo) + (Aq@Wgv + bgv) * sigmoid(Aq@Wgg + bgg)
// A tiles double-buffered; weight (B) tiles single-buffered (L2-resident).
__global__ __launch_bounds__(256)
void gemm3_glu_kernel(const _Float16* __restrict__ Ao,
                      const _Float16* __restrict__ Aq,
                      const _Float16* __restrict__ BoT,
                      const _Float16* __restrict__ BvT,
                      const _Float16* __restrict__ BgT,
                      const float* __restrict__ bo,
                      const float* __restrict__ bgv,
                      const float* __restrict__ bgg,
                      float* __restrict__ res, int Dout, int K) {
  __shared__ __align__(16) _Float16 As1[2][FTM * AST];
  __shared__ __align__(16) _Float16 As2[2][FTM * AST];
  __shared__ __align__(16) _Float16 Bs1[FTN * AST];
  __shared__ __align__(16) _Float16 Bs2[FTN * AST];
  __shared__ __align__(16) _Float16 Bs3[FTN * AST];

  const int tid  = threadIdx.x;
  const int wave = tid >> 5;
  const int lane = tid & 31;
  const int bm   = blockIdx.x * FTM;
  const int bn   = blockIdx.y * FTN;

  f8 accO[4], accV[4], accG[4];
#pragma unroll
  for (int j = 0; j < 4; ++j)
#pragma unroll
    for (int i = 0; i < 8; ++i) { accO[j][i] = 0.f; accV[j][i] = 0.f; accG[j][i] = 0.f; }

  const int arow = tid >> 1;          // 0..127
  const int acol = (tid & 1) * 16;
  const int brow = tid >> 2;          // 0..63
  const int bcol = (tid & 3) * 8;
  const int aOff = arow * AST + acol;
  const int bOff = brow * AST + bcol;
  const _Float16* aoBase = Ao + (long long)(bm + arow) * K + acol;
  const _Float16* aqBase = Aq + (long long)(bm + arow) * K + acol;
  const long long bBase  = (long long)(bn + brow) * K + bcol;

  // prologue: stage A tiles for k0=0 into buffer 0
  async_copy_b128(aoBase,     As1[0] + aOff);
  async_copy_b128(aoBase + 8, As1[0] + aOff + 8);
  async_copy_b128(aqBase,     As2[0] + aOff);
  async_copy_b128(aqBase + 8, As2[0] + aOff + 8);

  const int nk = K / TK;
  for (int t = 0; t < nk; ++t) {
    const int buf = t & 1;
    const int k0  = t * TK;
    // B tiles for this step (single-buffered)
    async_copy_b128(BoT + bBase + k0, Bs1 + bOff);
    async_copy_b128(BvT + bBase + k0, Bs2 + bOff);
    async_copy_b128(BgT + bBase + k0, Bs3 + bOff);
    if (t + 1 < nk) {
      const int k1 = k0 + TK;
      async_copy_b128(aoBase + k1,     As1[buf ^ 1] + aOff);
      async_copy_b128(aoBase + k1 + 8, As1[buf ^ 1] + aOff + 8);
      async_copy_b128(aqBase + k1,     As2[buf ^ 1] + aOff);
      async_copy_b128(aqBase + k1 + 8, As2[buf ^ 1] + aOff + 8);
      async_wait4();   // retires A(t) remainder + B(t); A(t+1) stays in flight
    } else {
      async_wait0();
    }
    __syncthreads();

    h16 a1 = make_a_frag(As1[buf], wave * 16, lane);
    h16 a2 = make_a_frag(As2[buf], wave * 16, lane);
#pragma unroll
    for (int j = 0; j < 4; ++j) {
      h16 b1 = make_b_frag(Bs1, j * 16, lane);
      accO[j] = __builtin_amdgcn_wmma_f32_16x16x32_f16(false, a1, false, b1, (short)0, accO[j], false, false);
      h16 b2 = make_b_frag(Bs2, j * 16, lane);
      accV[j] = __builtin_amdgcn_wmma_f32_16x16x32_f16(false, a2, false, b2, (short)0, accV[j], false, false);
      h16 b3 = make_b_frag(Bs3, j * 16, lane);
      accG[j] = __builtin_amdgcn_wmma_f32_16x16x32_f16(false, a2, false, b3, (short)0, accG[j], false, false);
    }
    __syncthreads();
  }

  const int colLane = lane & 15;
  const int mbase   = (lane >> 4) * 8;
#pragma unroll
  for (int j = 0; j < 4; ++j) {
    const int n = bn + j * 16 + colLane;
    const float cb0 = bo[n], cb1 = bgv[n], cb2 = bgg[n];
#pragma unroll
    for (int g = 0; g < 8; ++g) {
      const int m = bm + wave * 16 + mbase + g;
      const float o  = accO[j][g] + cb0;
      const float gv = accV[j][g] + cb1;
      const float gg = accG[j][g] + cb2;
      const float sig = 1.0f / (1.0f + __expf(-gg));
      res[(long long)m * Dout + n] = o + gv * sig;
    }
  }
}

// ---------------- elementwise / reduction kernels ---------------------------

// vectorized f32 -> f16 (8 elements per thread-step)
__global__ __launch_bounds__(256)
void cvt_f32_to_f16_kernel(const float* __restrict__ in, _Float16* __restrict__ out,
                           long long n8) {
  long long i = (long long)blockIdx.x * blockDim.x + threadIdx.x;
  long long stride = (long long)gridDim.x * blockDim.x;
  for (; i < n8; i += stride) {
    f4 v0 = *(const f4*)(in + i * 8);
    f4 v1 = *(const f4*)(in + i * 8 + 4);
    h8 o;
#pragma unroll
    for (int j = 0; j < 4; ++j) { o[j] = (_Float16)v0[j]; o[j + 4] = (_Float16)v1[j]; }
    *(h8*)(out + i * 8) = o;
  }
}

// Wt[e*D + d] = (f16) W[d*D + e]   (32x32 LDS tile transpose)
__global__ __launch_bounds__(256)
void w_transpose_f16_kernel(const float* __restrict__ W, _Float16* __restrict__ Wt, int D) {
  __shared__ float tile[32][33];
  const int bx = blockIdx.x * 32;
  const int by = blockIdx.y * 32;
  const int tx = threadIdx.x;       // 0..31
  const int ty = threadIdx.y;       // 0..7
#pragma unroll
  for (int r = 0; r < 4; ++r)
    tile[ty + r * 8][tx] = W[(long long)(by + ty + r * 8) * D + bx + tx];
  __syncthreads();
#pragma unroll
  for (int r = 0; r < 4; ++r)
    Wt[(long long)(bx + ty + r * 8) * D + by + tx] = (_Float16)tile[tx][ty + r * 8];
}

// vectorized softmax: ncols == 2048, 256 threads, one h8 chunk per thread
__global__ __launch_bounds__(256)
void softmax_rows_kernel(_Float16* __restrict__ P, int ncols) {
  const long long row = blockIdx.x;
  _Float16* p = P + row * (long long)ncols;
  __shared__ float red[256];
  const int tid = threadIdx.x;

  h8 v = *(const h8*)(p + tid * 8);
  float xv[8];
#pragma unroll
  for (int i = 0; i < 8; ++i) xv[i] = (float)v[i];

  float lmax = xv[0];
#pragma unroll
  for (int i = 1; i < 8; ++i) lmax = fmaxf(lmax, xv[i]);
  red[tid] = lmax; __syncthreads();
  for (int s = 128; s > 0; s >>= 1) { if (tid < s) red[tid] = fmaxf(red[tid], red[tid + s]); __syncthreads(); }
  const float m = red[0]; __syncthreads();

  float e[8], lsum = 0.0f;
#pragma unroll
  for (int i = 0; i < 8; ++i) { e[i] = __expf(xv[i] - m); lsum += e[i]; }
  red[tid] = lsum; __syncthreads();
  for (int s = 128; s > 0; s >>= 1) { if (tid < s) red[tid] += red[tid + s]; __syncthreads(); }
  const float inv = 1.0f / red[0];

  h8 o;
#pragma unroll
  for (int i = 0; i < 8; ++i) o[i] = (_Float16)(e[i] * inv);
  *(h8*)(p + tid * 8) = o;
}

// vectorized LayerNorm: D == 1024, 256 threads, one float4 per thread
__global__ __launch_bounds__(256)
void layernorm_kernel(const float* __restrict__ res,
                      const float* __restrict__ gamma,
                      const float* __restrict__ beta,
                      float* __restrict__ out, int D) {
  const long long row = blockIdx.x;
  const float* p = res + row * (long long)D;
  __shared__ float rs[256];
  __shared__ float rq[256];
  const int tid = threadIdx.x;

  f4 v = *(const f4*)(p + tid * 4);
  float s = v[0] + v[1] + v[2] + v[3];
  float q = v[0]*v[0] + v[1]*v[1] + v[2]*v[2] + v[3]*v[3];
  rs[tid] = s; rq[tid] = q; __syncthreads();
  for (int st = 128; st > 0; st >>= 1) {
    if (tid < st) { rs[tid] += rs[tid + st]; rq[tid] += rq[tid + st]; }
    __syncthreads();
  }
  const float mu  = rs[0] / D;
  const float var = rq[0] / D - mu * mu;
  const float inv = rsqrtf(var + 1e-5f);

  f4 g = *(const f4*)(gamma + tid * 4);
  f4 b = *(const f4*)(beta + tid * 4);
  f4 o;
#pragma unroll
  for (int i = 0; i < 4; ++i) o[i] = (v[i] - mu) * inv * g[i] + b[i];
  *(f4*)(out + row * (long long)D + tid * 4) = o;
}

// ---------------------------------------------------------------------------
// Host-side orchestration
// ---------------------------------------------------------------------------
extern "C" void kernel_launch(void* const* d_in, const int* in_sizes, int n_in,
                              void* d_out, int out_size, void* d_ws, size_t ws_size,
                              hipStream_t stream) {
  (void)in_sizes; (void)n_in; (void)out_size; (void)ws_size;
  constexpr int B = 4, S = 2048, D = 1024;
  constexpr long long BS = (long long)B * S;            // 8192

  const float* x    = (const float*)d_in[0];
  const float* Wq   = (const float*)d_in[1];
  const float* bq   = (const float*)d_in[2];
  const float* Wk   = (const float*)d_in[3];
  const float* bk   = (const float*)d_in[4];
  const float* Wv   = (const float*)d_in[5];
  const float* bv   = (const float*)d_in[6];
  const float* Wo   = (const float*)d_in[7];
  const float* bo   = (const float*)d_in[8];
  const float* Wgv  = (const float*)d_in[9];
  const float* bgv  = (const float*)d_in[10];
  const float* Wgg  = (const float*)d_in[11];
  const float* bgg  = (const float*)d_in[12];
  const float* gamma = (const float*)d_in[13];
  const float* beta  = (const float*)d_in[14];
  float* out = (float*)d_out;

  // workspace carve-up (256B aligned)
  char* w = (char*)d_ws;
  size_t off = 0;
  auto carve = [&](size_t bytes) -> void* {
    void* p = w + off;
    off = (off + bytes + 255) & ~(size_t)255;
    return p;
  };
  _Float16* xh  = (_Float16*)carve(BS * D * 2);
  _Float16* Wt[6];
  for (int i = 0; i < 6; ++i) Wt[i] = (_Float16*)carve((size_t)D * D * 2);
  _Float16* qh  = (_Float16*)carve(BS * D * 2);
  _Float16* kh  = (_Float16*)carve(BS * D * 2);
  _Float16* vht = (_Float16*)carve(BS * D * 2);                 // v transposed [b][e][s]
  _Float16* P   = (_Float16*)carve((size_t)B * S * S * 2);      // scores -> probs
  _Float16* oh  = (_Float16*)carve(BS * D * 2);
  float*    res = (float*)carve(BS * D * 4);

  const dim3 blk256(256);

  // 1) convert x -> f16
  cvt_f32_to_f16_kernel<<<4096, blk256, 0, stream>>>(x, xh, (BS * D) / 8);

  // 2) convert + transpose 6 weight matrices -> f16 [out x in]
  {
    const float* Ws[6] = {Wq, Wk, Wv, Wo, Wgv, Wgg};
    dim3 grid(D / 32, D / 32), blkT(32, 8);
    for (int i = 0; i < 6; ++i)
      w_transpose_f16_kernel<<<grid, blkT, 0, stream>>>(Ws[i], Wt[i], D);
  }

  // 3) q = x@Wq + bq ; k = x@Wk + bk   (f16 out, M = 8192)
  {
    dim3 grid(BS / GTM, D / GTN, 1);
    gemm_wmma_kernel<0, true><<<grid, blk256, 0, stream>>>(
        xh, D, 0, Wt[0], D, 0, qh, D, 0, bq, 1.0f, D);
    gemm_wmma_kernel<0, true><<<grid, blk256, 0, stream>>>(
        xh, D, 0, Wt[1], D, 0, kh, D, 0, bk, 1.0f, D);
  }

  // 4) v = x@Wv + bv, written transposed per batch: vht[b][e][s]
  {
    dim3 grid(S / GTM, D / GTN, B);
    gemm_wmma_kernel<1, true><<<grid, blk256, 0, stream>>>(
        xh, D, (long long)S * D, Wt[2], D, 0,
        vht, S, (long long)D * S, bv, 1.0f, D);
  }

  // 5) scores = (q @ k^T) / sqrt(D)  (batched; Bt = k row-major works directly)
  {
    dim3 grid(S / GTM, S / GTN, B);
    gemm_wmma_kernel<0, false><<<grid, blk256, 0, stream>>>(
        qh, D, (long long)S * D, kh, D, (long long)S * D,
        P, S, (long long)S * S, nullptr, 0.03125f /*1/sqrt(1024)*/, D);
  }

  // 6) softmax rows in place
  softmax_rows_kernel<<<B * S, blk256, 0, stream>>>(P, S);

  // 7) attn_out = P @ V  (Bt = vht, [e x s] per batch)
  {
    dim3 grid(S / GTM, D / GTN, B);
    gemm_wmma_kernel<0, false><<<grid, blk256, 0, stream>>>(
        P, S, (long long)S * S, vht, S, (long long)D * S,
        oh, D, (long long)S * D, nullptr, 1.0f, S);
  }

  // 8) fused: res = (oh@Wo + bo) + (qh@Wgv + bgv) * sigmoid(qh@Wgg + bgg)
  {
    dim3 grid(BS / FTM, D / FTN, 1);
    gemm3_glu_kernel<<<grid, blk256, 0, stream>>>(
        oh, qh, Wt[3], Wt[4], Wt[5], bo, bgv, bgg, res, D, D);
  }

  // 9) LayerNorm -> d_out (fp32)
  layernorm_kernel<<<BS, blk256, 0, stream>>>(res, gamma, beta, out, D);
}
